// ChannelAttention_11416023073078
// MI455X (gfx1250) — matrix operand order repaired and checked by reading, not compiled, not run on hardware
//
#include <hip/hip_runtime.h>
#include <hip/hip_bf16.h>

typedef __attribute__((ext_vector_type(16))) __bf16 v16bf;
typedef __attribute__((ext_vector_type(8)))  float  v8f;

#define Bn 8
#define Cc 512
#define NN 4096
#define NH 8
#define HD 64
#define C3 1536

// A-operand (16-bit 16xK), ISA 7.12.2: lanes 0-15 hold K=0..7/16..23, lanes 16-31 K=8..15/24..31
__device__ __forceinline__ int laneA(int row, int kk) {
  return (row & 15) + (((kk >> 3) & 1) << 4);
}
__device__ __forceinline__ int elemA(int kk) {
  return (kk & 7) + (((kk >> 4) & 1) << 3);
}
// B-operand (16-bit Kx16): lanes 0-15 hold K=0..7/16..23, lanes 16-31 K=8..15/24..31
__device__ __forceinline__ int laneB(int col, int kk) {
  return (col & 15) + (((kk >> 3) & 1) << 4);
}
__device__ __forceinline__ int elemB(int kk) { return (kk & 7) + (((kk >> 4) & 1) << 3); }

// Single-instruction packed fp32->bf16 pair conversion (RNE), as emitted by the
// compiler itself for bf16 stores: v_cvt_pk_bf16_f32 dst, lo, hi
__device__ __forceinline__ unsigned pack2bf(float lo, float hi) {
  unsigned u;
  asm("v_cvt_pk_bf16_f32 %0, %1, %2" : "=v"(u) : "v"(lo), "v"(hi));
  return u;
}

// Low 32 bits of a generic LDS pointer == LDS byte offset (aperture is in addr[63:32]).
__device__ __forceinline__ unsigned ldsOff(const void* p) {
  return (unsigned)(unsigned long long)p;
}

// CDNA5 async LDS<->global (ASYNCcnt-tracked, bypasses VGPRs)
__device__ __forceinline__ void asyncLoadB128(const void* g, unsigned lds) {
  asm volatile("global_load_async_to_lds_b128 %0, %1, off"
               :: "v"(lds), "v"((unsigned long long)g) : "memory");
}
__device__ __forceinline__ void asyncStoreB128(void* g, unsigned lds) {
  asm volatile("global_store_async_from_lds_b128 %0, %1, off"
               :: "v"((unsigned long long)g), "v"(lds) : "memory");
}
__device__ __forceinline__ void waitAsync0() {
  asm volatile("s_wait_asynccnt 0x0" ::: "memory");
}

// ---------------------------------------------------------------------------
// Kernel 1: qkv[b,n,j] = sum_c x[b,c,n] * w_qkv[c,j];  k-part scaled by 0.125
// 64(n) x 128(j) tile, 8 waves as 4x2; paired-K bf16 staging; output restaged
// through LDS and written with async b128 stores.
// ---------------------------------------------------------------------------
__global__ __launch_bounds__(256)
void qkv_gemm_kernel(const float* __restrict__ X, const float* __restrict__ Wqkv,
                     __bf16* __restrict__ qkvB) {
  __shared__ __align__(16) __bf16 aOp[4][32][16];
  __shared__ __align__(16) __bf16 bOp[8][32][16];
  __shared__ __align__(16) __bf16 stg[64][128];
  const int n0 = blockIdx.x * 64;
  const int j0 = blockIdx.y * 128;
  const int b  = blockIdx.z;
  const int t = threadIdx.x;
  const int lane = t & 31;
  const int w = t >> 5;
  const int wm = w & 3;
  const int wn = w >> 2;

  v8f acc[4] = {};

  for (int c0 = 0; c0 < Cc; c0 += 32) {
    // A (x^T tile): 64 rows(n) x 32 k(c); pack k-pairs -> one b32 LDS store
#pragma unroll
    for (int i = 0; i < 4; ++i) {
      int idx = t + 256 * i;           // 1024 pairs
      int m  = idx & 63;
      int k  = ((idx >> 6) & 15) * 2;
      const float* src = X + (((size_t)b * Cc + c0 + k) * NN) + n0 + m;
      unsigned v = pack2bf(src[0], src[NN]);
      *(unsigned*)(&aOp[m >> 4][laneA(m, k)][elemA(k)]) = v;
    }
    // B (w_qkv tile): 32 k(c) x 128 cols(j); pack k-pairs
#pragma unroll
    for (int i = 0; i < 8; ++i) {
      int idx = t + 256 * i;           // 2048 pairs
      int nn = idx & 127;
      int k  = ((idx >> 7) & 15) * 2;
      const float* src = Wqkv + ((size_t)(c0 + k)) * C3 + j0 + nn;
      unsigned v = pack2bf(src[0], src[C3]);
      *(unsigned*)(&bOp[nn >> 4][laneB(nn, k)][elemB(k)]) = v;
    }
    __syncthreads();
    v16bf a = *(const v16bf*)(&aOp[wm][lane][0]);
#pragma unroll
    for (int jt = 0; jt < 4; ++jt) {
      v16bf bb = *(const v16bf*)(&bOp[wn * 4 + jt][lane][0]);
      acc[jt] = __builtin_amdgcn_wmma_f32_16x16x32_bf16(
          false, a, false, bb, (short)0, acc[jt], false, false);
    }
    __syncthreads();
  }
  // restage C tile into LDS (bf16), then coalesced async b128 stores
#pragma unroll
  for (int jt = 0; jt < 4; ++jt) {
    int jl = wn * 64 + jt * 16 + (lane & 15);
    float mul = (((j0 + jl) >> 9) == 1) ? 0.125f : 1.0f;  // k-part scale hd^-0.5
#pragma unroll
    for (int r = 0; r < 8; ++r) {
      int ml = wm * 16 + r + ((lane >> 4) << 3);
      stg[ml][jl] = (__bf16)(acc[jt][r] * mul);
    }
  }
  __syncthreads();
#pragma unroll
  for (int i = 0; i < 4; ++i) {
    int gran = t + 256 * i;            // 1024 x 16B granules
    int row = gran >> 4;
    int cg  = gran & 15;
    __bf16* gdst = qkvB + ((size_t)b * NN + n0 + row) * C3 + j0 + cg * 8;
    asyncStoreB128(gdst, ldsOff(&stg[0][0]) + gran * 16);
  }
  waitAsync0();
}

__global__ void zero_f32_kernel(float* p, int n) {
  int i = blockIdx.x * blockDim.x + threadIdx.x;
  if (i < n) p[i] = 0.0f;
}

// ---------------------------------------------------------------------------
// Kernel 2: attn[bh,d,e] += sum_{n in quarter} k[n,d] * v[n,e]
// (K dimension is strided in memory -> manual staging; fp32 global atomics)
// ---------------------------------------------------------------------------
__global__ __launch_bounds__(128)
void ktv_gemm_kernel(const __bf16* __restrict__ qkvB, float* __restrict__ attnF) {
  __shared__ __align__(16) __bf16 aOp[4][32][16];  // k^T tiles: rows d, K = n
  __shared__ __align__(16) __bf16 bOp[4][32][16];  // v tiles:   K = n, cols e
  const int quarter = blockIdx.x;
  const int h = blockIdx.y;
  const int b = blockIdx.z;
  const int t = threadIdx.x;
  const int lane = t & 31;
  const int s = t >> 5;

  v8f acc[4] = {};
  for (int it = 0; it < 32; ++it) {
    const int n0 = quarter * 1024 + it * 32;
#pragma unroll
    for (int i = 0; i < 16; ++i) {
      int idx = t + 128 * i;
      int d  = idx & 63;
      int nn = idx >> 6;  // 0..31
      size_t rb = ((size_t)b * NN + n0 + nn) * C3 + h * HD + d;
      aOp[d >> 4][laneA(d, nn)][elemA(nn)] = qkvB[rb + Cc];      // k (pre-scaled)
      bOp[d >> 4][laneB(d, nn)][elemB(nn)] = qkvB[rb + 2 * Cc];  // v
    }
    __syncthreads();
    v16bf a = *(const v16bf*)(&aOp[s][lane][0]);
#pragma unroll
    for (int et = 0; et < 4; ++et) {
      v16bf bb = *(const v16bf*)(&bOp[et][lane][0]);
      acc[et] = __builtin_amdgcn_wmma_f32_16x16x32_bf16(
          false, a, false, bb, (short)0, acc[et], false, false);
    }
    __syncthreads();
  }
  const int bh = b * NH + h;
#pragma unroll
  for (int et = 0; et < 4; ++et) {
    int e = et * 16 + (lane & 15);
#pragma unroll
    for (int r = 0; r < 8; ++r) {
      int d = s * 16 + r + ((lane >> 4) << 3);
      unsafeAtomicAdd(&attnF[((size_t)bh * HD + d) * HD + e], acc[et][r]);
    }
  }
}

// ---------------------------------------------------------------------------
// Kernel 3: row softmax of attn, emit bf16 in B-operand order (attn^T: K=e, col=d)
// ---------------------------------------------------------------------------
__global__ __launch_bounds__(64)
void softmax_kernel(const float* __restrict__ attnF, __bf16* __restrict__ attnOp) {
  const int bh = blockIdx.x;
  const int d = threadIdx.x;
  const float* row = attnF + ((size_t)bh * HD + d) * HD;
  float m = -3.402823466e38f;
  for (int e = 0; e < 64; ++e) m = fmaxf(m, row[e]);
  float ex[64];
  float s = 0.0f;
  for (int e = 0; e < 64; ++e) { ex[e] = __expf(row[e] - m); s += ex[e]; }
  float inv = 1.0f / s;
  for (int e = 0; e < 64; ++e) {
    int kt = e >> 5, kk = e & 31;
    attnOp[(((size_t)bh * 2 + kt) * 4 + (d >> 4)) * 512 + laneB(d, kk) * 16 + elemB(kk)] =
        (__bf16)(ex[e] * inv);
  }
}

// ---------------------------------------------------------------------------
// Kernel 4: outh[b,n,h*64+d] = sum_e attn[d,e] * q[n,e]  (= q @ attn^T)
// q staged via async b128 loads directly into A-operand layout (K is contiguous
// in memory -> each operand half-row is a 16B contiguous granule). Output is
// restaged in LDS (reusing aOp) and written with async b128 stores.
// ---------------------------------------------------------------------------
__global__ __launch_bounds__(256)
void attn_q_kernel(const __bf16* __restrict__ qkvB, const __bf16* __restrict__ attnOp,
                   __bf16* __restrict__ outh) {
  __shared__ __align__(16) __bf16 aOp[16][2][32][16];  // 32 KB; reused as 256x64 store stage
  const int n0 = blockIdx.x * 256;
  const int h = blockIdx.y;
  const int b = blockIdx.z;
  const int t = threadIdx.x;
  const int lane = t & 31;
  const int w = t >> 5;
  const int bh = b * NH + h;

  // async-stage q tile 256(n) x 64(e) into operand order: 2048 16B granules
#pragma unroll
  for (int i = 0; i < 8; ++i) {
    int g = t + 256 * i;
    int eh = g & 1;
    int gl = (g >> 1) & 31;
    int kt = (g >> 6) & 1;
    int mt = g >> 7;
    int nn = mt * 16 + (gl & 15);
    int e = kt * 32 + eh * 16 + (gl >> 4) * 8;
    const __bf16* gsrc = qkvB + ((size_t)b * NN + n0 + nn) * C3 + h * HD + e;
    asyncLoadB128(gsrc, ldsOff(&aOp[mt][kt][gl][eh * 8]));
  }

  v16bf bfrag[2][4];
#pragma unroll
  for (int kt = 0; kt < 2; ++kt)
#pragma unroll
    for (int dt = 0; dt < 4; ++dt)
      bfrag[kt][dt] = *(const v16bf*)(attnOp +
          (((size_t)bh * 2 + kt) * 4 + dt) * 512 + lane * 16);

  waitAsync0();
  __syncthreads();

  v8f acc[2][4] = {};
#pragma unroll
  for (int ms = 0; ms < 2; ++ms) {
    int mt = w * 2 + ms;
#pragma unroll
    for (int kt = 0; kt < 2; ++kt) {
      v16bf a = *(const v16bf*)(&aOp[mt][kt][lane][0]);
#pragma unroll
      for (int dt = 0; dt < 4; ++dt)
        acc[ms][dt] = __builtin_amdgcn_wmma_f32_16x16x32_bf16(
            false, a, false, bfrag[kt][dt], (short)0, acc[ms][dt], false, false);
    }
  }
  __syncthreads();  // all waves done reading aOp; reuse it as store stage
  __bf16* stg = &aOp[0][0][0][0];  // [256][64]
#pragma unroll
  for (int ms = 0; ms < 2; ++ms) {
    int mt = w * 2 + ms;
#pragma unroll
    for (int dt = 0; dt < 4; ++dt) {
      int d = dt * 16 + (lane & 15);
#pragma unroll
      for (int r = 0; r < 8; ++r) {
        int nl = mt * 16 + r + ((lane >> 4) << 3);
        stg[nl * 64 + d] = (__bf16)acc[ms][dt][r];
      }
    }
  }
  __syncthreads();
#pragma unroll
  for (int i = 0; i < 8; ++i) {
    int gran = t + 256 * i;            // 2048 x 16B granules
    int row = gran >> 3;
    int cg  = gran & 7;
    __bf16* gdst = outh + ((size_t)b * NN + n0 + row) * Cc + h * HD + cg * 8;
    asyncStoreB128(gdst, ldsOff(stg) + gran * 16);
  }
  waitAsync0();
}

// ---------------------------------------------------------------------------
// Kernel 5: out[b,co,n] = sum_c w_proj[c,co] * outh[b,n,c] + bias[co]
// A = w_proj^T (paired-K fp32->bf16 staging); B = outh^T staged with async
// b128 loads (K contiguous in memory). N on lanes -> coalesced fp32 stores.
// ---------------------------------------------------------------------------
__global__ __launch_bounds__(256)
void proj_gemm_kernel(const __bf16* __restrict__ outh, const float* __restrict__ Wp,
                      const float* __restrict__ bias, float* __restrict__ out) {
  __shared__ __align__(16) __bf16 aOp[4][32][16];  // w_proj^T: 64 rows(co) x 32 k(c)
  __shared__ __align__(16) __bf16 bOp[8][32][16];  // outh^T:   32 k(c) x 128 cols(n)
  const int n0  = blockIdx.x * 128;
  const int co0 = blockIdx.y * 64;
  const int b = blockIdx.z;
  const int t = threadIdx.x;
  const int lane = t & 31;
  const int w = t >> 5;
  const int wm = w & 3;
  const int wn = w >> 2;

  v8f acc[4] = {};
  for (int c0 = 0; c0 < Cc; c0 += 32) {
    // async-stage B: 512 x 16B granules (8 contiguous c per granule)
#pragma unroll
    for (int i = 0; i < 2; ++i) {
      int g = t + 256 * i;
      int eh = g & 1;
      int gl = (g >> 1) & 31;
      int nt = g >> 6;
      int nn = nt * 16 + (gl & 15);
      int c = c0 + (gl >> 4) * 8 + eh * 16;
      const __bf16* gsrc = outh + ((size_t)b * NN + n0 + nn) * Cc + c;
      asyncLoadB128(gsrc, ldsOff(&bOp[nt][gl][eh * 8]));
    }
    // manual A staging with k-pair packing (fp32 source)
#pragma unroll
    for (int i = 0; i < 4; ++i) {
      int idx = t + 256 * i;           // 1024 pairs
      int m = idx & 63;
      int k = ((idx >> 6) & 15) * 2;
      const float* src = Wp + ((size_t)(c0 + k)) * Cc + co0 + m;
      unsigned v = pack2bf(src[0], src[Cc]);
      *(unsigned*)(&aOp[m >> 4][laneA(m, k)][elemA(k)]) = v;
    }
    waitAsync0();
    __syncthreads();
    v16bf a = *(const v16bf*)(&aOp[wm][lane][0]);
#pragma unroll
    for (int jt = 0; jt < 4; ++jt) {
      v16bf bb = *(const v16bf*)(&bOp[wn * 4 + jt][lane][0]);
      acc[jt] = __builtin_amdgcn_wmma_f32_16x16x32_bf16(
          false, a, false, bb, (short)0, acc[jt], false, false);
    }
    __syncthreads();
  }
#pragma unroll
  for (int r = 0; r < 8; ++r) {
    int co = co0 + wm * 16 + r + ((lane >> 4) << 3);
    float bv = bias[co];
#pragma unroll
    for (int jt = 0; jt < 4; ++jt) {
      int n = n0 + wn * 64 + jt * 16 + (lane & 15);
      out[((size_t)b * Cc + co) * NN + n] = acc[jt][r] + bv;
    }
  }
}

extern "C" void kernel_launch(void* const* d_in, const int* in_sizes, int n_in,
                              void* d_out, int out_size, void* d_ws, size_t ws_size,
                              hipStream_t stream) {
  (void)in_sizes; (void)n_in; (void)out_size; (void)ws_size;
  const float* x      = (const float*)d_in[0];
  const float* w_qkv  = (const float*)d_in[1];
  const float* w_proj = (const float*)d_in[2];
  const float* b_proj = (const float*)d_in[3];
  // d_in[4] = num_heads (fixed 8, compiled in)

  char* ws = (char*)d_ws;
  __bf16* qkvB   = (__bf16*)(ws);                                 // 96 MB
  float*  attnF  = (float*) (ws + 100663296);                     // 1 MB
  __bf16* attnOp = (__bf16*)(ws + 100663296 + 1048576);           // 0.5 MB
  __bf16* outh   = (__bf16*)(ws + 100663296 + 1048576 + 524288);  // 32 MB
  float* out = (float*)d_out;

  qkv_gemm_kernel<<<dim3(NN / 64, C3 / 128, Bn), 256, 0, stream>>>(x, w_qkv, qkvB);
  zero_f32_kernel<<<dim3((Bn * NH * HD * HD + 255) / 256), 256, 0, stream>>>(
      attnF, Bn * NH * HD * HD);
  ktv_gemm_kernel<<<dim3(4, NH, Bn), 128, 0, stream>>>(qkvB, attnF);
  softmax_kernel<<<dim3(Bn * NH), 64, 0, stream>>>(attnF, attnOp);
  attn_q_kernel<<<dim3(NN / 256, NH, Bn), 256, 0, stream>>>(qkvB, attnOp, outh);
  proj_gemm_kernel<<<dim3(NN / 128, Cc / 64, Bn), 256, 0, stream>>>(
      outh, w_proj, b_proj, out);
}
